// DynamicGraphReservoir_16767552324177
// MI455X (gfx1250) — matrix operand order repaired and checked by reading, not compiled, not run on hardware
//
#include <hip/hip_runtime.h>
#include <hip/hip_bf16.h>

// Problem constants (from the reference):
//   T=8, N=20000, F=128, H=256, E=320000, leak=0.9 (device scalar)
//
// All matmul/graph operands live in bf16 (reservoir + tanh is precision-tolerant);
// the f32 master state is kept for the leaky accumulation across steps.
// Workspace layout (d_ws):
//   s0_f32[N*H f32] | s0_bf[N*H bf16] | s1_bf[N*H bf16] | agg_bf[N*H bf16]
//   | x_bf[T*N*F bf16] | w_in0_bf | w_rec0_bf | w_in1_bf | w_rec1_bf   (~93 MB)

typedef __bf16 bf16x4  __attribute__((ext_vector_type(4)));
typedef __bf16 bf16x8  __attribute__((ext_vector_type(8)));
typedef __bf16 bf16x16 __attribute__((ext_vector_type(16)));
typedef float  f32x8   __attribute__((ext_vector_type(8)));
typedef int    v4i     __attribute__((vector_size(16)));   // matches builtin param type

constexpr int HDIM = 256;   // hidden width (N dim of GEMMs)
constexpr int BM   = 128;   // macro tile rows
constexpr int BN   = 64;    // macro tile cols
constexpr int BK   = 32;    // K chunk (one bf16 WMMA K)
constexpr int LDA  = BK + 8; // padded LDS stride (halfwords) -> 80B rows, 16B aligned
constexpr int LDW  = BK + 8;

#if __has_builtin(__builtin_amdgcn_global_load_async_to_lds_b128) && \
    __has_builtin(__builtin_amdgcn_s_wait_asynccnt)
#define ASYNC_LDS 1
#else
#define ASYNC_LDS 0
#endif

__device__ __forceinline__ float fast_tanh(float x) {
#if __has_builtin(__builtin_amdgcn_tanhf)
  return __builtin_amdgcn_tanhf(x);     // V_TANH_F32 (CDNA5 trans op)
#else
  return tanhf(x);
#endif
}

// CDNA5 packed bf16 atomic add (non-returning): GLOBAL_ATOMIC_PK_ADD_BF16.
__device__ __forceinline__ void atomic_pk_add_bf16(unsigned int* p, unsigned int d) {
  asm volatile("global_atomic_pk_add_bf16 %0, %1, off" :: "v"(p), "v"(d) : "memory");
}

// 16B global -> LDS copy: async data-mover path (no VGPR round trip) or sync fallback.
__device__ __forceinline__ void copy16_to_lds(const __bf16* gp, unsigned short* lp) {
#if ASYNC_LDS
  __builtin_amdgcn_global_load_async_to_lds_b128((v4i*)gp, (v4i*)lp, 0, 0);
#else
  *reinterpret_cast<uint4*>(lp) = *reinterpret_cast<const uint4*>(gp);
#endif
}

template <int N>
__device__ __forceinline__ void wait_async() {
#if ASYNC_LDS
  __builtin_amdgcn_s_wait_asynccnt(N);
#endif
}

// ---------------- zero fill (16B granules) ----------------
__global__ void zero_kernel(float* __restrict__ p, long n4) {
  long i = (long)blockIdx.x * blockDim.x + threadIdx.x;
  if (i < n4) reinterpret_cast<float4*>(p)[i] = make_float4(0.f, 0.f, 0.f, 0.f);
}

// ---------------- f32 -> bf16 streaming convert (x, weights; once per launch) ----
__global__ void cvt_kernel(const float* __restrict__ s, __bf16* __restrict__ d, long n4) {
  long i = (long)blockIdx.x * blockDim.x + threadIdx.x;
  if (i < n4) {
    float4 v = reinterpret_cast<const float4*>(s)[i];
    bf16x4 p = { (__bf16)v.x, (__bf16)v.y, (__bf16)v.z, (__bf16)v.w };
    reinterpret_cast<bf16x4*>(d)[i] = p;
  }
}

// ---------------- edge scatter-add: agg[dst] += state[src], all bf16 ------------
// 32 threads per edge; each thread moves 16B (8 bf16) and issues 4 pk atomics.
__global__ __launch_bounds__(256)
void scatter_kernel(const __bf16* __restrict__ state,
                    const int* __restrict__ src,
                    const int* __restrict__ dst,
                    __bf16* __restrict__ agg, int E) {
  int t = blockIdx.x * blockDim.x + threadIdx.x;
  int e = t >> 5;
  if (e >= E) return;
  int c = (t & 31) << 3;
  uint4 v = *reinterpret_cast<const uint4*>(state + (size_t)src[e] * HDIM + c);
  unsigned int* out = reinterpret_cast<unsigned int*>(agg + (size_t)dst[e] * HDIM + c);
  atomic_pk_add_bf16(out + 0, v.x);
  atomic_pk_add_bf16(out + 1, v.y);
  atomic_pk_add_bf16(out + 2, v.z);
  atomic_pk_add_bf16(out + 3, v.w);
}

// ---------------- one K-chunk of WMMAs for this wave's 32x32 region ----------------
__device__ __forceinline__ void mma_step(const unsigned short* __restrict__ lds_a,
                                         const unsigned short* __restrict__ lds_w,
                                         int lane, int wm, int wn, f32x8 (&acc)[2][2]) {
  const int mrow = lane & 15;
  const int hs   = (lane >> 4) & 1;   // half-wave select
  bf16x16 afr[2], bfr[2];
#pragma unroll
  for (int mi = 0; mi < 2; ++mi) {
    // A 16x32 bf16 fragment: lane holds row (l%16); K = hs*8 + {0..7} and hs*8 + {16..23}
    const unsigned short* p = lds_a + (wm * 32 + mi * 16 + mrow) * LDA + hs * 8;
    bf16x8 lo = *reinterpret_cast<const bf16x8*>(p);
    bf16x8 hi = *reinterpret_cast<const bf16x8*>(p + 16);
    afr[mi] = __builtin_shufflevector(lo, hi, 0,1,2,3,4,5,6,7,8,9,10,11,12,13,14,15);
  }
#pragma unroll
  for (int ni = 0; ni < 2; ++ni) {
    // B 32x16 bf16 fragment: lane holds col (l%16); K = hs*8 + {0..7} and hs*8 + {16..23}; B[k][n] = W[n][k]
    const unsigned short* p = lds_w + (wn * 32 + ni * 16 + mrow) * LDW + hs * 8;
    bf16x8 lo = *reinterpret_cast<const bf16x8*>(p);
    bf16x8 hi = *reinterpret_cast<const bf16x8*>(p + 16);
    bfr[ni] = __builtin_shufflevector(lo, hi, 0,1,2,3,4,5,6,7,8,9,10,11,12,13,14,15);
  }
#pragma unroll
  for (int mi = 0; mi < 2; ++mi)
#pragma unroll
    for (int ni = 0; ni < 2; ++ni)
      acc[mi][ni] = __builtin_amdgcn_wmma_f32_16x16x32_bf16(
          false, afr[mi], false, bfr[ni], (short)0, acc[mi][ni], false, false);
}

// ---------------- fused reservoir update ----------------
// s_new = leak * tanh(A @ Win^T + G @ Wrec^T) + (1-leak) * s_old  (f32 master)
// plus a bf16 mirror of s_new for the next graph/matmul consumers.
// Double-buffered LDS pipeline: tile k+1 streams into LDS via the async data
// mover while tile k's WMMAs execute. Each tile issue is exactly 3 async b128
// instructions per wave, so s_wait_asynccnt(3) retires the previous tile.
__global__ __launch_bounds__(256)
void reservoir_gemm_kernel(const __bf16* __restrict__ A, int K1,
                           const __bf16* __restrict__ G,
                           const __bf16* __restrict__ Win,
                           const __bf16* __restrict__ Wrec,
                           const float* __restrict__ s_old,
                           float* __restrict__ s_new,
                           __bf16* __restrict__ s_new_bf,
                           int nrows, const float* __restrict__ leak_ptr) {
  __shared__ __align__(16) unsigned short lds_a[2][BM * LDA];
  __shared__ __align__(16) unsigned short lds_w[2][BN * LDW];

  const int tid  = threadIdx.x;
  const int lane = tid & 31;
  const int wid  = tid >> 5;       // 8 waves
  const int wm   = wid & 3;        // 4 waves along M
  const int wn   = wid >> 2;       // 2 waves along N
  const int m0   = blockIdx.x * BM;
  const int n0   = blockIdx.y * BN;

  const int n1    = K1 / BK;
  const int total = n1 + HDIM / BK;

  // Per-thread staging coordinates (fixed across chunks).
  const int ra  = (tid + 0)   >> 2;            // A-tile rows for chunk pair
  const int rb  = (tid + 256) >> 2;
  const int c8  = (tid & 3) << 3;              // 16B column offset within tile
  const int rw  = tid >> 2;                    // W-tile row

  auto issue_tile = [&](int buf, int kc) {
    const __bf16* a; const __bf16* wsrc; int ld, k0;
    if (kc < n1) { a = A; wsrc = Win;  ld = K1;   k0 = kc * BK; }
    else         { a = G; wsrc = Wrec; ld = HDIM; k0 = (kc - n1) * BK; }
    int ga = m0 + ra; if (ga >= nrows) ga = nrows - 1;   // clamp (stores guarded)
    int gb = m0 + rb; if (gb >= nrows) gb = nrows - 1;
    copy16_to_lds(a + (size_t)ga * ld + k0 + c8, &lds_a[buf][ra * LDA + c8]);
    copy16_to_lds(a + (size_t)gb * ld + k0 + c8, &lds_a[buf][rb * LDA + c8]);
    copy16_to_lds(wsrc + (size_t)(n0 + rw) * ld + k0 + c8, &lds_w[buf][rw * LDW + c8]);
  };

  f32x8 acc[2][2] = {};

  issue_tile(0, 0);
  for (int kc = 0; kc < total; ++kc) {
    const int cur = kc & 1;
    if (kc + 1 < total) {
      issue_tile(cur ^ 1, kc + 1);   // prefetch next tile into the other buffer
      wait_async<3>();               // previous 3 async ops (this tile) retired
    } else {
      wait_async<0>();
    }
    __syncthreads();                 // all waves' LDS writes for `cur` visible
    mma_step(lds_a[cur], lds_w[cur], lane, wm, wn, acc);
    __syncthreads();                 // done reading `cur` before it is reused
  }

  // Epilogue: tanh + leaky mix -> f32 master + bf16 mirror
  const float leak = *leak_ptr;
  const float keep = 1.0f - leak;
  const int mrow = lane & 15;
  const int hs   = (lane >> 4) & 1;
#pragma unroll
  for (int mi = 0; mi < 2; ++mi) {
#pragma unroll
    for (int ni = 0; ni < 2; ++ni) {
      const int col = n0 + wn * 32 + ni * 16 + mrow;
#pragma unroll
      for (int v = 0; v < 8; ++v) {
        const int row = m0 + wm * 32 + mi * 16 + hs * 8 + v;
        if (row < nrows) {
          const size_t o = (size_t)row * HDIM + col;
          float val = leak * fast_tanh(acc[mi][ni][v]) + keep * s_old[o];
          s_new[o] = val;
          s_new_bf[o] = (__bf16)val;
        }
      }
    }
  }
}

extern "C" void kernel_launch(void* const* d_in, const int* in_sizes, int n_in,
                              void* d_out, int out_size, void* d_ws, size_t ws_size,
                              hipStream_t stream) {
  constexpr int T = 8, NN = 20000, FDIM = 128;
  const float* x      = (const float*)d_in[0];
  const int*   eidx   = (const int*)d_in[1];
  const float* w_in0  = (const float*)d_in[2];
  const float* w_rec0 = (const float*)d_in[3];
  const float* w_in1  = (const float*)d_in[4];
  const float* w_rec1 = (const float*)d_in[5];
  const float* leak   = (const float*)d_in[6];
  const int E = in_sizes[1] / 2;
  const int* src = eidx;
  const int* dst = eidx + E;

  const size_t nstate = (size_t)NN * HDIM;      // 5,120,000
  float* s1 = (float*)d_out;                    // layer-1 f32 state lives in d_out

  char* w = (char*)d_ws;
  float*  s0      = (float*)w;   w += nstate * 4;
  __bf16* s0_bf   = (__bf16*)w;  w += nstate * 2;
  __bf16* s1_bf   = (__bf16*)w;  w += nstate * 2;
  __bf16* agg_bf  = (__bf16*)w;  w += nstate * 2;
  __bf16* x_bf    = (__bf16*)w;  w += (size_t)T * NN * FDIM * 2;
  __bf16* win0_bf = (__bf16*)w;  w += (size_t)HDIM * FDIM * 2;
  __bf16* wrec0_bf= (__bf16*)w;  w += (size_t)HDIM * HDIM * 2;
  __bf16* win1_bf = (__bf16*)w;  w += (size_t)HDIM * HDIM * 2;
  __bf16* wrec1_bf= (__bf16*)w;

  auto blocks4 = [](long n4) { return (int)((n4 + 255) / 256); };

  // one-time f32 -> bf16 conversions
  const long xq = (long)T * NN * FDIM / 4;
  cvt_kernel<<<blocks4(xq), 256, 0, stream>>>(x, x_bf, xq);
  cvt_kernel<<<blocks4(HDIM*FDIM/4), 256, 0, stream>>>(w_in0,  win0_bf,  HDIM*FDIM/4);
  cvt_kernel<<<blocks4(HDIM*HDIM/4), 256, 0, stream>>>(w_rec0, wrec0_bf, HDIM*HDIM/4);
  cvt_kernel<<<blocks4(HDIM*HDIM/4), 256, 0, stream>>>(w_in1,  win1_bf,  HDIM*HDIM/4);
  cvt_kernel<<<blocks4(HDIM*HDIM/4), 256, 0, stream>>>(w_rec1, wrec1_bf, HDIM*HDIM/4);

  // zero initial states (f32 masters + bf16 mirrors)
  const long zf = (long)(nstate / 4);     // f32 buffers, 16B granules
  const long zb = (long)(nstate / 8);     // bf16 buffers, 16B granules
  zero_kernel<<<blocks4(zf), 256, 0, stream>>>(s0, zf);
  zero_kernel<<<blocks4(zf), 256, 0, stream>>>(s1, zf);
  zero_kernel<<<blocks4(zb), 256, 0, stream>>>((float*)s0_bf, zb);
  zero_kernel<<<blocks4(zb), 256, 0, stream>>>((float*)s1_bf, zb);

  const int sblocks = (int)(((long)E * 32 + 255) / 256);
  dim3 ggrid((NN + BM - 1) / BM, HDIM / BN);   // 157 x 4

  for (int t = 0; t < T; ++t) {
    const __bf16* xt = x_bf + (size_t)t * NN * FDIM;
    // layer 0
    zero_kernel<<<blocks4(zb), 256, 0, stream>>>((float*)agg_bf, zb);
    scatter_kernel<<<sblocks, 256, 0, stream>>>(s0_bf, src, dst, agg_bf, E);
    reservoir_gemm_kernel<<<ggrid, 256, 0, stream>>>(xt, FDIM, agg_bf, win0_bf, wrec0_bf,
                                                     s0, s0, s0_bf, NN, leak);
    // layer 1
    zero_kernel<<<blocks4(zb), 256, 0, stream>>>((float*)agg_bf, zb);
    scatter_kernel<<<sblocks, 256, 0, stream>>>(s1_bf, src, dst, agg_bf, E);
    reservoir_gemm_kernel<<<ggrid, 256, 0, stream>>>(s0_bf, HDIM, agg_bf, win1_bf, wrec1_bf,
                                                     s1, s1, s1_bf, NN, leak);
  }
}
